// RelativeMultiHeadAttnPro_66597762892436
// MI455X (gfx1250) — matrix operand order repaired and checked by reading, not compiled, not run on hardware
//
#include <hip/hip_runtime.h>
#include <hip/hip_bf16.h>
#include <stdint.h>

#define SEQ    512
#define DMODEL 1024
#define NH     16
#define HD     64
#define BATCH  8
#define TWO_L  1024
#define N3D    3072

typedef __attribute__((ext_vector_type(16))) __bf16 v16bf;
typedef __attribute__((ext_vector_type(8)))  __bf16 v8bf;
typedef __attribute__((ext_vector_type(8)))  float  v8f;
typedef __attribute__((ext_vector_type(4)))  unsigned int u32x4;
typedef __attribute__((ext_vector_type(8)))  int    i32x8;
typedef __attribute__((ext_vector_type(4)))  int    i32x4;

// ---------------- helpers ----------------

__device__ __forceinline__ __bf16 f2bf(float f) {
  union { float f; unsigned u; } cv; cv.f = f;
  unsigned u = cv.u + 0x7fffu + ((cv.u >> 16) & 1u);   // round-to-nearest-even
  union { unsigned short s; __bf16 b; } r; r.s = (unsigned short)(u >> 16);
  return r.b;
}

// 2D tile load Global -> LDS via Tensor Data Mover.
// tile_w/tile_h/row_stride in elements; ds_code: 1 = 2B (bf16), 2 = 4B (f32).
// Issued once by the calling wave (EXEC ignored for tensor ops). Caller pairs
// with s_wait_tensorcnt + workgroup barrier.
__device__ __forceinline__ void tdm_load_2d(const void* gsrc, const void* lds_dst,
                                            int tile_w, int tile_h, int row_stride,
                                            int ds_code) {
  unsigned long long ga = (unsigned long long)(size_t)gsrc;
  u32x4 g0;
  g0[0] = 1u;                                             // count=1, user D#
  g0[1] = (unsigned)(size_t)lds_dst;                      // lds_addr (bytes)
  g0[2] = (unsigned)(ga & 0xffffffffull);                 // global_addr[31:0]
  g0[3] = (unsigned)((ga >> 32) & 0x1ffffffull) | (2u << 30); // addr[56:32] | type=2
  const unsigned tw = (unsigned)row_stride;               // tensor_dim0
  const unsigned th = 1u << 20;                           // tensor_dim1 (large; tiles in-bounds)
  i32x8 g1;
  g1[0] = (int)((unsigned)ds_code << 16);                 // wg_mask=0, data_size
  g1[1] = (int)((tw & 0xffffu) << 16);                    // tensor_dim0[15:0]
  g1[2] = (int)((tw >> 16) | ((th & 0xffffu) << 16));     // dim0[31:16] | dim1[15:0]
  g1[3] = (int)((th >> 16) | ((unsigned)tile_w << 16));   // dim1[31:16] | tile_dim0
  g1[4] = (int)((unsigned)tile_h & 0xffffu);              // tile_dim1, tile_dim2=0
  g1[5] = (int)row_stride;                                // tensor_dim0_stride[31:0]
  g1[6] = 0;
  g1[7] = 0;
  i32x4 z4 = {0, 0, 0, 0};
#if defined(__clang_major__) && (__clang_major__ >= 23)
  i32x8 z8 = {0, 0, 0, 0, 0, 0, 0, 0};
  __builtin_amdgcn_tensor_load_to_lds(g0, g1, z4, z4, z8, 0);
#else
  __builtin_amdgcn_tensor_load_to_lds(g0, g1, z4, z4, 0);
#endif
}

// A fragment (16x32 bf16, M row per lane) from row-major LDS [rows][32].
__device__ __forceinline__ v16bf frag_a_lds(const __bf16* sA, int rowBase) {
  const int lane = threadIdx.x & 31;
  const __bf16* p = sA + (rowBase + (lane & 15)) * 32 + ((lane >> 4) << 3);
  v8bf lo = *(const v8bf*)(p);        // K 0..7   (or 8..15)
  v8bf hi = *(const v8bf*)(p + 16);   // K 16..23 (or 24..31)
  return __builtin_shufflevector(lo, hi, 0,1,2,3,4,5,6,7,8,9,10,11,12,13,14,15);
}

// B fragment (32x16 bf16, N col per lane) from LDS stored as [N][K=32].
__device__ __forceinline__ v16bf frag_b_lds(const __bf16* sBT, int colBase) {
  const int lane = threadIdx.x & 31;
  const __bf16* p = sBT + (colBase + (lane & 15)) * 32 + ((lane >> 4) << 3);
  v8bf lo = *(const v8bf*)(p);
  v8bf hi = *(const v8bf*)(p + 16);
  return __builtin_shufflevector(lo, hi, 0,1,2,3,4,5,6,7,8,9,10,11,12,13,14,15);  // K 0..7,16..23 (lanes 0-15) / 8..15,24..31 (lanes 16-31)
}

__device__ __forceinline__ v8f wmma_bf16(v16bf a, v16bf b, v8f c) {
  return __builtin_amdgcn_wmma_f32_16x16x32_bf16(false, a, false, b, (short)0, c, false, false);
}

// ---------------- K0: rel_posT_bf[p][d] = bf16(rel_pos[d][p]) ----------------

__global__ __launch_bounds__(256) void relpos_transpose_kernel(
    const float* __restrict__ rel_pos, __bf16* __restrict__ relT) {
  const int i = blockIdx.x * 256 + threadIdx.x;   // over 1024*64
  const int d = i & 63, p = i >> 6;
  relT[i] = f2bf(rel_pos[d * TWO_L + p]);
}

// ---------------- K1: QKV = x @ Wqkv (f32 in, bf16 operands out) ----------------
// Pipelined: TDM for tile it+1 overlaps convert+WMMA of tile it.

__global__ __launch_bounds__(256) void qkv_gemm_kernel(
    const float* __restrict__ x, const float* __restrict__ Wqkv,
    const float* __restrict__ r_r_bias, const float* __restrict__ r_w_bias,
    __bf16* __restrict__ q_rr_bf, __bf16* __restrict__ q_rw_bf,
    __bf16* __restrict__ k_bf, __bf16* __restrict__ vT_bf) {
  __shared__ __align__(16) float  stA[2][128 * 32];
  __shared__ __align__(16) float  stB[2][32 * 128];
  __shared__ __align__(32) __bf16 sA [2][128 * 32];
  __shared__ __align__(32) __bf16 sBT[2][128 * 32];

  const int mBase = blockIdx.x * 128;
  const int nBase = blockIdx.y * 128;
  const int tid  = threadIdx.x;
  const int wave = tid >> 5;
  const int lane = tid & 31;
  const int wm = wave >> 1, wn = wave & 1;

  v8f acc[2][4];
  const v8f vz = {0.f,0.f,0.f,0.f,0.f,0.f,0.f,0.f};
  for (int r = 0; r < 2; ++r) for (int c = 0; c < 4; ++c) acc[r][c] = vz;

  if (wave == 0) {
    tdm_load_2d(x    + (size_t)mBase * DMODEL, stA[0], 32, 128, DMODEL, 2);
    tdm_load_2d(Wqkv + (size_t)nBase,          stB[0], 128, 32, N3D,    2);
  }

  for (int it = 0; it < 32; ++it) {
    const int p = it & 1;
    if (wave == 0) __builtin_amdgcn_s_wait_tensorcnt(0);
    __syncthreads();
    if (wave == 0 && it + 1 < 32) {
      const int kn = (it + 1) * 32;
      tdm_load_2d(x    + (size_t)mBase * DMODEL + kn, stA[p ^ 1], 32, 128, DMODEL, 2);
      tdm_load_2d(Wqkv + (size_t)kn * N3D + nBase,    stB[p ^ 1], 128, 32, N3D,    2);
    }
    for (int i = tid; i < 128 * 32; i += 256) sA[p][i] = f2bf(stA[p][i]);
    for (int i = tid; i < 32 * 128; i += 256) {
      int kk = i >> 7, n = i & 127;
      sBT[p][n * 32 + kk] = f2bf(stB[p][i]);
    }
    __syncthreads();
    v16bf af[2], bfr[4];
    for (int r = 0; r < 2; ++r) af[r]  = frag_a_lds(sA[p],  wm * 32 + r * 16);
    for (int c = 0; c < 4; ++c) bfr[c] = frag_b_lds(sBT[p], wn * 64 + c * 16);
    for (int r = 0; r < 2; ++r)
      for (int c = 0; c < 4; ++c)
        acc[r][c] = wmma_bf16(af[r], bfr[c], acc[r][c]);
  }

  // epilogue: 64-wide column strip -> exactly one of {q,k,v} and one head.
  // q is emitted twice with the two biases pre-folded; v is emitted transposed.
  const int nWave = nBase + wn * 64;
  const int sel = nWave >> 10;
  const int h   = (nWave >> 6) & 15;
  const int mo = (lane >> 4) << 3;
  const int ln = lane & 15;
  for (int r = 0; r < 2; ++r)
    for (int c = 0; c < 4; ++c) {
      const int dd = c * 16 + ln;
      const float brr = r_r_bias[h * HD + dd];
      const float brw = r_w_bias[h * HD + dd];
      for (int rr = 0; rr < 8; ++rr) {
        const int m = mBase + wm * 32 + r * 16 + mo + rr;
        const int b = m >> 9, l = m & 511;
        const float a = acc[r][c][rr];
        const size_t idx = (((size_t)b * NH + h) * SEQ + l) * HD + dd;
        if (sel == 0)      { q_rr_bf[idx] = f2bf(a + brr); q_rw_bf[idx] = f2bf(a + brw); }
        else if (sel == 1) { k_bf[idx] = f2bf(a); }
        else               { vT_bf[(((size_t)b * NH + h) * HD + dd) * SEQ + l] = f2bf(a); }
      }
    }
}

// ---------------- K2: BD = (q+r_w) @ rel_pos ; E = k @ rel_pos (bf16 direct) ----------------

__global__ __launch_bounds__(256) void relpos_gemm_kernel(
    const __bf16* __restrict__ q_rw_bf, const __bf16* __restrict__ k_bf,
    const __bf16* __restrict__ relT_bf,
    float* __restrict__ BD_ws, float* __restrict__ E_ws) {
  __shared__ __align__(32) __bf16 sA [2][128 * 32];
  __shared__ __align__(32) __bf16 sBT[2][128 * 32];

  const int mBase = blockIdx.x * 128;       // over b*h*L = 65536
  const int nBase = blockIdx.y * 128;       // over 2L = 1024
  const int which = blockIdx.z;             // 0 = BD, 1 = E
  const __bf16* Asrc = which ? k_bf : q_rw_bf;
  float* dst = which ? E_ws : BD_ws;

  const int tid = threadIdx.x, wave = tid >> 5, lane = tid & 31;
  const int wm = wave >> 1, wn = wave & 1;

  v8f acc[2][4];
  const v8f vz = {0.f,0.f,0.f,0.f,0.f,0.f,0.f,0.f};
  for (int r = 0; r < 2; ++r) for (int c = 0; c < 4; ++c) acc[r][c] = vz;

  if (wave == 0) {
    tdm_load_2d(Asrc    + (size_t)mBase * HD, sA[0],  32, 128, HD, 1);
    tdm_load_2d(relT_bf + (size_t)nBase * HD, sBT[0], 32, 128, HD, 1);
  }
  for (int it = 0; it < 2; ++it) {
    const int p = it & 1;
    if (wave == 0) __builtin_amdgcn_s_wait_tensorcnt(0);
    __syncthreads();
    if (wave == 0 && it == 0) {
      tdm_load_2d(Asrc    + (size_t)mBase * HD + 32, sA[1],  32, 128, HD, 1);
      tdm_load_2d(relT_bf + (size_t)nBase * HD + 32, sBT[1], 32, 128, HD, 1);
    }
    v16bf af[2], bfr[4];
    for (int r = 0; r < 2; ++r) af[r]  = frag_a_lds(sA[p],  wm * 32 + r * 16);
    for (int c = 0; c < 4; ++c) bfr[c] = frag_b_lds(sBT[p], wn * 64 + c * 16);
    for (int r = 0; r < 2; ++r)
      for (int c = 0; c < 4; ++c)
        acc[r][c] = wmma_bf16(af[r], bfr[c], acc[r][c]);
  }

  const int mo = (lane >> 4) << 3, ln = lane & 15;
  for (int r = 0; r < 2; ++r)
    for (int c = 0; c < 4; ++c)
      for (int rr = 0; rr < 8; ++rr) {
        const int m = mBase + wm * 32 + r * 16 + mo + rr;
        const int n = nBase + wn * 64 + c * 16 + ln;
        dst[(size_t)m * TWO_L + n] = acc[r][c][rr];
      }
}

// ---------------- K3: scores = (q+r_r)k^T/8 + BDshift/8 + Eshift/8 + prev ----------------

__global__ __launch_bounds__(256) void score_kernel(
    const __bf16* __restrict__ q_rr_bf, const __bf16* __restrict__ k_bf,
    const float* __restrict__ BD_ws, const float* __restrict__ E_ws,
    const float* __restrict__ prev, const int* __restrict__ skip_op,
    float* __restrict__ prev_out) {
  __shared__ __align__(32) __bf16 sA [2][128 * 32];
  __shared__ __align__(32) __bf16 sBT[2][128 * 32];

  const int z = blockIdx.z;                 // b*16 + h
  const int iBase = blockIdx.x * 128;
  const int jBase = blockIdx.y * 128;
  const int tid = threadIdx.x, wave = tid >> 5, lane = tid & 31;
  const int wm = wave >> 1, wn = wave & 1;

  v8f acc[2][4];
  const v8f vz = {0.f,0.f,0.f,0.f,0.f,0.f,0.f,0.f};
  for (int r = 0; r < 2; ++r) for (int c = 0; c < 4; ++c) acc[r][c] = vz;

  const __bf16* Abase = q_rr_bf + ((size_t)z * SEQ + iBase) * HD;
  const __bf16* Bbase = k_bf    + ((size_t)z * SEQ + jBase) * HD;   // k rows are [N][K]
  if (wave == 0) {
    tdm_load_2d(Abase, sA[0],  32, 128, HD, 1);
    tdm_load_2d(Bbase, sBT[0], 32, 128, HD, 1);
  }
  for (int it = 0; it < 2; ++it) {
    const int p = it & 1;
    if (wave == 0) __builtin_amdgcn_s_wait_tensorcnt(0);
    __syncthreads();
    if (wave == 0 && it == 0) {
      tdm_load_2d(Abase + 32, sA[1],  32, 128, HD, 1);
      tdm_load_2d(Bbase + 32, sBT[1], 32, 128, HD, 1);
    }
    v16bf af[2], bfr[4];
    for (int r = 0; r < 2; ++r) af[r]  = frag_a_lds(sA[p],  wm * 32 + r * 16);
    for (int c = 0; c < 4; ++c) bfr[c] = frag_b_lds(sBT[p], wn * 64 + c * 16);
    for (int r = 0; r < 2; ++r)
      for (int c = 0; c < 4; ++c)
        acc[r][c] = wmma_bf16(af[r], bfr[c], acc[r][c]);
  }

  const int mo = (lane >> 4) << 3, ln = lane & 15;
  const int sk = *skip_op;
  for (int r = 0; r < 2; ++r)
    for (int c = 0; c < 4; ++c)
      for (int rr = 0; rr < 8; ++rr) {
        const int i = iBase + wm * 32 + r * 16 + mo + rr;
        const int j = jBase + wn * 64 + c * 16 + ln;
        float val = acc[r][c][rr];
        val += BD_ws[((size_t)z * SEQ + i) * TWO_L + (SEQ + j - i)];   // shifted BD
        val += E_ws [((size_t)z * SEQ + j) * TWO_L + (SEQ + i - j)];   // shifted E^T
        val *= 0.125f;                                                 // 1/sqrt(64)
        if (sk) val += prev[((size_t)z * SEQ + i) * SEQ + j];
        prev_out[((size_t)z * SEQ + i) * SEQ + j] = val;
      }
}

// ---------------- K4: row softmax, emit bf16 attn (one wave per row) ----------------

__global__ __launch_bounds__(256) void softmax_kernel(
    const float* __restrict__ prev_out, __bf16* __restrict__ attn_bf) {
  const int row  = blockIdx.x * 8 + (threadIdx.x >> 5);
  const int lane = threadIdx.x & 31;
  const float* src = prev_out + (size_t)row * SEQ;
  float v[16];
  float m = -3.4e38f;
  for (int t = 0; t < 16; ++t) { v[t] = src[lane + 32 * t]; m = fmaxf(m, v[t]); }
  for (int s = 16; s; s >>= 1) m = fmaxf(m, __shfl_xor(m, s, 32));
  float sum = 0.f;
  for (int t = 0; t < 16; ++t) { v[t] = __expf(v[t] - m); sum += v[t]; }
  for (int s = 16; s; s >>= 1) sum += __shfl_xor(sum, s, 32);
  const float inv = 1.f / sum;
  __bf16* dst = attn_bf + (size_t)row * SEQ;
  for (int t = 0; t < 16; ++t) dst[lane + 32 * t] = f2bf(v[t] * inv);
}

// ---------------- K5: out = attn @ v (bf16 direct), re-interleave heads ----------------

__global__ __launch_bounds__(256) void pv_kernel(
    const __bf16* __restrict__ attn_bf, const __bf16* __restrict__ vT_bf,
    float* __restrict__ out) {
  __shared__ __align__(32) __bf16 sA [2][128 * 32];
  __shared__ __align__(32) __bf16 sBT[2][64 * 32];

  const int z = blockIdx.z;
  const int b = z >> 4, h = z & 15;
  const int iBase = blockIdx.x * 128;
  const int tid = threadIdx.x, wave = tid >> 5, lane = tid & 31;

  v8f acc[4];
  const v8f vz = {0.f,0.f,0.f,0.f,0.f,0.f,0.f,0.f};
  for (int c = 0; c < 4; ++c) acc[c] = vz;

  const __bf16* Abase = attn_bf + ((size_t)z * SEQ + iBase) * SEQ;
  const __bf16* Bbase = vT_bf   + (size_t)z * HD * SEQ;              // [64][512]
  if (wave == 0) {
    tdm_load_2d(Abase, sA[0],  32, 128, SEQ, 1);
    tdm_load_2d(Bbase, sBT[0], 32, 64,  SEQ, 1);
  }
  for (int it = 0; it < 16; ++it) {
    const int p = it & 1;
    if (wave == 0) __builtin_amdgcn_s_wait_tensorcnt(0);
    __syncthreads();
    if (wave == 0 && it + 1 < 16) {
      const int kn = (it + 1) * 32;
      tdm_load_2d(Abase + kn, sA[p ^ 1],  32, 128, SEQ, 1);
      tdm_load_2d(Bbase + kn, sBT[p ^ 1], 32, 64,  SEQ, 1);
    }
    v16bf af = frag_a_lds(sA[p], wave * 16);
    for (int c = 0; c < 4; ++c)
      acc[c] = wmma_bf16(af, frag_b_lds(sBT[p], c * 16), acc[c]);
  }

  const int mo = (lane >> 4) << 3, ln = lane & 15;
  for (int c = 0; c < 4; ++c)
    for (int rr = 0; rr < 8; ++rr) {
      const int i = iBase + wave * 16 + mo + rr;
      const int n = c * 16 + ln;
      out[((size_t)b * SEQ + i) * DMODEL + h * HD + n] = acc[c][rr];
    }
}

// ---------------- launch ----------------

extern "C" void kernel_launch(void* const* d_in, const int* in_sizes, int n_in,
                              void* d_out, int out_size, void* d_ws, size_t ws_size,
                              hipStream_t stream) {
  const float* x        = (const float*)d_in[0];
  const float* prev     = (const float*)d_in[1];
  const float* Wqkv     = (const float*)d_in[2];
  const float* r_r_bias = (const float*)d_in[3];
  const float* r_w_bias = (const float*)d_in[4];
  const float* rel_pos  = (const float*)d_in[5];
  const int*   skip_op  = (const int*)d_in[6];

  float* out      = (float*)d_out;                               // (8,512,1024)
  float* prev_out = out + (size_t)BATCH * SEQ * DMODEL;          // (8,16,512,512)

  const size_t qkv_e = (size_t)BATCH * NH * SEQ * HD;            // 4 Mi elems
  char* w = (char*)d_ws;
  __bf16* q_rr_bf = (__bf16*)w;
  __bf16* q_rw_bf = q_rr_bf + qkv_e;
  __bf16* k_bf    = q_rw_bf + qkv_e;
  __bf16* vT_bf   = k_bf    + qkv_e;
  __bf16* relT_bf = vT_bf   + qkv_e;                             // 64 Ki elems
  float*  BD_ws   = (float*)(w + (4 * qkv_e + (size_t)TWO_L * HD) * sizeof(__bf16));
  float*  E_ws    = BD_ws + (size_t)BATCH * NH * SEQ * TWO_L;
  __bf16* attn_bf = (__bf16*)BD_ws;   // reuse BD region once scores are written

  relpos_transpose_kernel<<<dim3(256),      256, 0, stream>>>(rel_pos, relT_bf);
  qkv_gemm_kernel        <<<dim3(32, 24),   256, 0, stream>>>(x, Wqkv, r_r_bias, r_w_bias,
                                                              q_rr_bf, q_rw_bf, k_bf, vT_bf);
  relpos_gemm_kernel     <<<dim3(512, 8, 2),256, 0, stream>>>(q_rw_bf, k_bf, relT_bf, BD_ws, E_ws);
  score_kernel           <<<dim3(4, 4, 128),256, 0, stream>>>(q_rr_bf, k_bf, BD_ws, E_ws,
                                                              prev, skip_op, prev_out);
  softmax_kernel         <<<dim3(8192),     256, 0, stream>>>(prev_out, attn_bf);
  pv_kernel              <<<dim3(4, 1, 128),256, 0, stream>>>(attn_bf, vT_bf, out);
}
